// LightningIndexer_27101243638247
// MI455X (gfx1250) — matrix operand order repaired and checked by reading, not compiled, not run on hardware
//
#include <hip/hip_runtime.h>
#include <hip/hip_bf16.h>

typedef __attribute__((ext_vector_type(16))) _Float16 v16h;
typedef __attribute__((ext_vector_type(8)))  _Float16 v8h;
typedef __attribute__((ext_vector_type(8)))  float    v8f;
typedef __attribute__((ext_vector_type(4)))  float    v4f;

#define D_MODEL 2048
#define N_HEADS 4
#define IDX_DIM 64
#define BATCH   2
#define TQ      4096
#define TK      4096
#define NQ_EXT  272              // 256 q_I cols + 4 w_I cols + 12 zero pad (17 n-tiles)
#define NT_QW   17               // NQ_EXT / 16
#define NT_K    4                // IDX_DIM / 16
#define KT      (D_MODEL / 32)   // 64 K-tiles of 32
#define M_ROWS  (BATCH * TQ)     // 8192 flat rows for both x_q and x_k

// ---------------------------------------------------------------------------
// Stage 1: convert weights to f16 AND pre-swizzle into WMMA B-fragment order.
//
// Fragment layout (16-bit B 32x16, wave32):
//   frag index f = (kt * n_tiles + nt)
//   lane: col n = nt*16 + (lane&15); khalf = lane>>4
//   halves 0..7 = K[kt*32 + khalf*8 .. +7], halves 8..15 = K[kt*32 + 16 + khalf*8 .. +7]
// Stored as frag-major / lane-major / half-major: lane's 16 halves are 32
// contiguous bytes -> proj_gemm loads each fragment with two b128s.
// Packed q-weights: logical cols [0,256)=Wq, [256,260)=Ww, [260,272)=0.
// ---------------------------------------------------------------------------
__global__ __launch_bounds__(256) void prep_weights(
    const float* __restrict__ Wq, const float* __restrict__ Ww,
    const float* __restrict__ Wk, _Float16* __restrict__ Wqw_f,
    _Float16* __restrict__ Wk_f) {
  int idx = blockIdx.x * blockDim.x + threadIdx.x;
  const int nqw = KT * NT_QW * 32 * 16;      // 557,056 halves
  const int nk  = KT * NT_K * 32 * 16;       // 131,072 halves
  if (idx < nqw) {
    int e    = idx & 15;
    int lane = (idx >> 4) & 31;
    int frag = idx >> 9;
    int nt   = frag % NT_QW;
    int kt   = frag / NT_QW;
    int col  = nt * 16 + (lane & 15);
    int k    = kt * 32 + (lane >> 4) * 8 + (e >> 3) * 16 + (e & 7);
    float v = 0.0f;
    if (col < N_HEADS * IDX_DIM) v = Wq[k * (N_HEADS * IDX_DIM) + col];
    else if (col < N_HEADS * IDX_DIM + N_HEADS)
      v = Ww[k * N_HEADS + (col - N_HEADS * IDX_DIM)];
    Wqw_f[idx] = (_Float16)v;
  } else if (idx < nqw + nk) {
    int j    = idx - nqw;
    int e    = j & 15;
    int lane = (j >> 4) & 31;
    int frag = j >> 9;
    int nt   = frag % NT_K;
    int kt   = frag / NT_K;
    int col  = nt * 16 + (lane & 15);
    int k    = kt * 32 + (lane >> 4) * 8 + (e >> 3) * 16 + (e & 7);
    Wk_f[j] = (_Float16)Wk[k * IDX_DIM + col];
  }
}

// ---------------------------------------------------------------------------
// Stage 2: projection GEMM  Y(f16, MxN) = X(f32, Mx2048) * W(frag-packed f16)
// One wave per 16x16 output tile; 64 chained v_wmma_f32_16x16x32_f16 steps.
// A: two aligned 32B f32 runs per lane (ISA 16-bit A layout), cvt to f16.
// B: fragment-packed -> two contiguous b128 loads per K-step.
// ---------------------------------------------------------------------------
__global__ __launch_bounds__(128) void proj_gemm(
    const float* __restrict__ X, const _Float16* __restrict__ Wf,
    _Float16* __restrict__ Y, int N, int n_tiles) {
  const int lane = threadIdx.x & 31;
  const int wave = threadIdx.x >> 5;
  const int nt   = blockIdx.x * 4 + wave;
  if (nt >= n_tiles) return;          // uniform per wave: EXEC stays all-1s
  const int n0 = nt * 16;
  const int m0 = blockIdx.y * 16;
  const int m  = lane & 15;
  const int hi = lane >> 4;

  const float*    xrow = X + (size_t)(m0 + m) * D_MODEL;
  const _Float16* wbase = Wf + ((size_t)nt * 32 + lane) * 16;
  const size_t    wstep = (size_t)n_tiles * 32 * 16;   // halves per k-tile
  v8f acc = {};

  for (int kt = 0; kt < KT; ++kt) {
    const int k0 = kt * 32;
    // --- A fragment: two 32B runs, f32 -> f16 in registers ---
    v4f a0 = *(const v4f*)(xrow + k0 + hi * 8);
    v4f a1 = *(const v4f*)(xrow + k0 + hi * 8 + 4);
    v4f a2 = *(const v4f*)(xrow + k0 + 16 + hi * 8);
    v4f a3 = *(const v4f*)(xrow + k0 + 16 + hi * 8 + 4);
    v16h a;
#pragma unroll
    for (int i = 0; i < 4; ++i) {
      a[i]      = (_Float16)a0[i];
      a[4 + i]  = (_Float16)a1[i];
      a[8 + i]  = (_Float16)a2[i];
      a[12 + i] = (_Float16)a3[i];
    }
    // --- B fragment: 32 contiguous bytes (2 x b128) ---
    const _Float16* wp = wbase + (size_t)kt * wstep;
    v8h b0 = *(const v8h*)(wp);
    v8h b1 = *(const v8h*)(wp + 8);
    v16h b;
#pragma unroll
    for (int i = 0; i < 8; ++i) { b[i] = b0[i]; b[8 + i] = b1[i]; }

    acc = __builtin_amdgcn_wmma_f32_16x16x32_f16(false, a, false, b,
                                                 (short)0, acc, false, false);
  }

  // C/D layout: element r -> row m0 + r + 8*hi, col n0 + (lane&15)
  _Float16* yp = Y + (size_t)(m0 + hi * 8) * N + n0 + m;
#pragma unroll
  for (int r = 0; r < 8; ++r) yp[(size_t)r * N] = (_Float16)acc[r];
}

// ---------------------------------------------------------------------------
// Stage 3: fused score kernel (HBM-store bound):
//   out[b,q,k] = sum_h relu( q_I[b,q,h,:] . k_I[b,k,:] ) * w_I[b,q,h]
// Each wave: one 16-row q tile. A fragments for all 4 heads (K=64 -> 2 chunks)
// and per-row head weights loaded ONCE, then stream L2-resident k tiles:
// 4 b128 B loads + 8 WMMAs + ReLU/scale in VALU + 64B-coalesced f32 stores.
// ---------------------------------------------------------------------------
__global__ __launch_bounds__(256) void score_kernel(
    const _Float16* __restrict__ qI, const _Float16* __restrict__ kI,
    float* __restrict__ out) {
  const int lane = threadIdx.x & 31;
  const int wave = threadIdx.x >> 5;
  const int m    = lane & 15;
  const int hi   = lane >> 4;
  const int b    = blockIdx.z;
  const int q0   = blockIdx.y * 16;
  const int gq   = b * TQ + q0;      // flat q-row base

  // A fragments: 4 heads x 2 K-chunks (b128 runs of 8 halves each)
  v16h A[N_HEADS][2];
  const _Float16* qrow = qI + (size_t)(gq + m) * NQ_EXT;
#pragma unroll
  for (int h = 0; h < N_HEADS; ++h) {
#pragma unroll
    for (int c = 0; c < 2; ++c) {
      v8h r0 = *(const v8h*)(qrow + h * IDX_DIM + c * 32 + hi * 8);
      v8h r1 = *(const v8h*)(qrow + h * IDX_DIM + c * 32 + 16 + hi * 8);
      v16h a;
#pragma unroll
      for (int i = 0; i < 8; ++i) { a[i] = r0[i]; a[8 + i] = r1[i]; }
      A[h][c] = a;
    }
  }

  // Per-lane head weights for the 8 output rows this lane owns (r + 8*hi)
  float wr[8][N_HEADS];
#pragma unroll
  for (int r = 0; r < 8; ++r) {
    const _Float16* wp =
        qI + (size_t)(gq + r + hi * 8) * NQ_EXT + N_HEADS * IDX_DIM;
#pragma unroll
    for (int h = 0; h < N_HEADS; ++h) wr[r][h] = (float)wp[h];
  }

  const _Float16* kbase = kI + (size_t)(b * TK + m) * IDX_DIM;
  float* obase = out + ((size_t)b * TQ + q0 + hi * 8) * TK + m;

  const int waveSlot = blockIdx.x * 8 + wave;  // 64 slots over 256 k-tiles
  for (int kt = waveSlot; kt < TK / 16; kt += 64) {
    const int n0 = kt * 16;
    // B fragments: lane n=lane&15 is k-token n0+n; two 16B runs per chunk (k = hi*8.., 16+hi*8..)
    const _Float16* krow = kbase + (size_t)n0 * IDX_DIM;
    v16h Bv[2];
#pragma unroll
    for (int c = 0; c < 2; ++c) {
      v8h r0 = *(const v8h*)(krow + c * 32 + hi * 8);
      v8h r1 = *(const v8h*)(krow + c * 32 + 16 + hi * 8);
      v16h bb;
#pragma unroll
      for (int i = 0; i < 8; ++i) { bb[i] = r0[i]; bb[8 + i] = r1[i]; }
      Bv[c] = bb;
    }

    v8f acc = {};
#pragma unroll
    for (int h = 0; h < N_HEADS; ++h) {
      v8f d = {};
      d = __builtin_amdgcn_wmma_f32_16x16x32_f16(false, A[h][0], false, Bv[0],
                                                 (short)0, d, false, false);
      d = __builtin_amdgcn_wmma_f32_16x16x32_f16(false, A[h][1], false, Bv[1],
                                                 (short)0, d, false, false);
#pragma unroll
      for (int r = 0; r < 8; ++r) {
        float v = d[r];
        v = v > 0.0f ? v : 0.0f;     // ReLU
        acc[r] += v * wr[r][h];      // head-weighted sum
      }
    }

    float* orow = obase + n0;
#pragma unroll
    for (int r = 0; r < 8; ++r) orow[(size_t)r * TK] = acc[r];
  }
}

// ---------------------------------------------------------------------------
// Workspace layout (bytes):
//   [0)            Wqw_f : 2048*272*2 = 1,114,112   (B-fragment packed)
//   [1,114,112)    Wk_f  : 2048* 64*2 =   262,144   (B-fragment packed)
//   [1,376,256)    qI    : 8192*272*2 = 4,456,448
//   [5,832,704)    kI    : 8192* 64*2 = 1,048,576   (total ~6.9 MB)
// ---------------------------------------------------------------------------
extern "C" void kernel_launch(void* const* d_in, const int* in_sizes, int n_in,
                              void* d_out, int out_size, void* d_ws,
                              size_t ws_size, hipStream_t stream) {
  const float* x_q = (const float*)d_in[0];
  const float* x_k = (const float*)d_in[1];
  const float* Wq  = (const float*)d_in[2];
  const float* Ww  = (const float*)d_in[3];
  const float* Wk  = (const float*)d_in[4];
  float* out = (float*)d_out;

  char* ws = (char*)d_ws;
  _Float16* Wqw_f = (_Float16*)(ws);
  _Float16* Wk_f  = (_Float16*)(ws + 1114112);
  _Float16* qIbuf = (_Float16*)(ws + 1376256);
  _Float16* kIbuf = (_Float16*)(ws + 5832704);

  // Stage 1: weight conversion + B-fragment swizzle
  const int prep_n = KT * NT_QW * 512 + KT * NT_K * 512;
  prep_weights<<<(prep_n + 255) / 256, 256, 0, stream>>>(Wq, Ww, Wk, Wqw_f,
                                                         Wk_f);

  // Stage 2: projections (q_I + w_I fused via packed weights; k_I)
  dim3 gq((NT_QW + 3) / 4, M_ROWS / 16);              // (5, 512)
  proj_gemm<<<gq, 128, 0, stream>>>(x_q, Wqw_f, qIbuf, NQ_EXT, NT_QW);
  dim3 gk(1, M_ROWS / 16);                            // (1, 512), 4 n-tiles
  proj_gemm<<<gk, 128, 0, stream>>>(x_k, Wk_f, kIbuf, IDX_DIM, NT_K);

  // Stage 3: fused scores -> d_out
  score_kernel<<<dim3(8, TQ / 16, BATCH), 256, 0, stream>>>(qIbuf, kIbuf, out);
}
